// LFM2AudioDetokenizer_63634235457784
// MI455X (gfx1250) — matrix operand order repaired and checked by reading, not compiled, not run on hardware
//
#include <hip/hip_runtime.h>
#include <hip/hip_bf16.h>
#include <math.h>

typedef __attribute__((ext_vector_type(16))) _Float16 v16h;
typedef __attribute__((ext_vector_type(8)))  _Float16 v8h;
typedef __attribute__((ext_vector_type(8)))  float    v8f;

#define D_MODEL 512
#define NHEAD   8
#define DHEAD   64
#define FFDIM   2048
#define NLAYER  4
#define CBOOK   8
#define VOCAB   2048
#define BATCH   2
#define TLEN    256
#define UPF     6
#define SLEN    (TLEN*UPF)      // 1536
#define MROWS   (BATCH*SLEN)    // 3072
#define SWINDOW 30
#define NFFTC   1280
#define HOPC    320
#define NBINS   641
#define YC      1282
#define KFFT    1312            // 1282 padded up to multiple of 32
#define NLIN    1344            // 1282 padded up to multiple of 64
#define OUTLEN  491520          // per-batch final samples
#define PADC    480
#define LDH     2048            // row stride of f16 activation buffers
#define LDQKV   1536            // row stride of fused qkv f32 buffer

// ---------------------------------------------------------------- weights f32[K,N] -> f16[Npad,K] (transposed, zero padded)
__global__ void k_convert_t(const float* __restrict__ src, _Float16* __restrict__ dst,
                            int K, int N, int Npad) {
  int idx = blockIdx.x*blockDim.x + threadIdx.x;
  if (idx >= Npad*K) return;
  int n = idx / K, k = idx - n*K;
  float v = (n < N) ? src[(size_t)k*N + n] : 0.0f;
  dst[(size_t)n*K + k] = (_Float16)v;
}

// ---------------------------------------------------------------- windowed inverse-rDFT table, [n=0..1279][c=0..1311], B-transposed
__global__ void k_build_tbl(_Float16* __restrict__ tbl) {
  int idx = blockIdx.x*blockDim.x + threadIdx.x;
  if (idx >= NFFTC*KFFT) return;
  int n = idx / KFFT, c = idx - n*KFFT;
  const float invN  = 1.0f/(float)NFFTC;
  const float twopi = 6.28318530717958647692f;
  float win = 0.5f*(1.0f - cosf(twopi*(float)n*invN));
  float v = 0.0f;
  if (c == 0) {                       // DC real
    v = invN;
  } else if (c < 640) {               // real bins 1..639
    int ph = (c*n) % NFFTC;
    v = 2.0f*invN*cosf(twopi*(float)ph*invN);
  } else if (c == 640) {              // Nyquist real
    v = (n & 1) ? -invN : invN;
  } else if (c >= 642 && c < 1281) {  // imag bins 1..639 (641 = imag DC = 0)
    int k = c - NBINS;
    int ph = (k*n) % NFFTC;
    v = -2.0f*invN*sinf(twopi*(float)ph*invN);
  }
  tbl[(size_t)n*KFFT + c] = (_Float16)(v*win);
}

// ---------------------------------------------------------------- embedding mean over codebooks + 6x nearest upsample
__global__ void k_embed(const int* __restrict__ codes, const float* __restrict__ emb,
                        float* __restrict__ x) {
  int bt = blockIdx.x;                 // [0, B*T)
  int b = bt / TLEN, t = bt - b*TLEN;
  int d0 = threadIdx.x*2;
  float a0 = 0.0f, a1 = 0.0f;
  #pragma unroll
  for (int cb = 0; cb < CBOOK; ++cb) {
    int id = codes[((size_t)b*CBOOK + cb)*TLEN + t] + cb*VOCAB;
    const float* row = emb + (size_t)id*D_MODEL;
    a0 += row[d0]; a1 += row[d0+1];
  }
  a0 *= (1.0f/CBOOK); a1 *= (1.0f/CBOOK);
  #pragma unroll
  for (int u = 0; u < UPF; ++u) {
    size_t r = ((size_t)b*SLEN + t*UPF + u)*D_MODEL;
    x[r + d0] = a0; x[r + d0 + 1] = a1;
  }
}

// ---------------------------------------------------------------- RMSNorm: f32 x-row -> f16 h-row (wave per row)
__global__ void k_rmsnorm(const float* __restrict__ x, const float* __restrict__ w,
                          _Float16* __restrict__ h) {
  int row  = blockIdx.x*(blockDim.x >> 5) + (threadIdx.x >> 5);
  int lane = threadIdx.x & 31;
  if (row >= MROWS) return;
  const float* xr = x + (size_t)row*D_MODEL;
  float v[16]; float ss = 0.0f;
  #pragma unroll
  for (int i = 0; i < 16; ++i) { v[i] = xr[lane + 32*i]; ss += v[i]*v[i]; }
  #pragma unroll
  for (int o = 16; o > 0; o >>= 1) ss += __shfl_xor(ss, o, 32);
  float sc = rsqrtf(ss*(1.0f/D_MODEL) + 1e-5f);
  _Float16* hr = h + (size_t)row*LDH;
  #pragma unroll
  for (int i = 0; i < 16; ++i) { int c = lane + 32*i; hr[c] = (_Float16)(v[i]*sc*w[c]); }
}

// ---------------------------------------------------------------- WMMA GEMM: C[M,N] = A[M,K](f16,lda) @ Bt[N,K](f16)
// wave computes a 32x64 strip: 2 A-fragments x 4 B-fragments = 8 WMMAs per
// K-step, 12 b128 loads (B reused x2). EPI: 0=store, 1=residual add,
// 2=+bias, 3=silu(aux)*acc -> f16 C16
template<int EPI>
__global__ void __launch_bounds__(128) k_gemm(
    const _Float16* __restrict__ A, const _Float16* __restrict__ Bt,
    float* __restrict__ C, _Float16* __restrict__ C16,
    const float* __restrict__ aux,
    int K, int lda, int nTiles, int Nout, int ldc) {
  int wid = blockIdx.x*(blockDim.x >> 5) + (threadIdx.x >> 5);
  if (wid >= (MROWS/32)*nTiles) return;
  int mt = wid / nTiles, nt = wid - mt*nTiles;
  int lane = threadIdx.x & 31;
  int hi   = lane >> 4;                         // 0 or 1
  // A fragments: lane holds rows mt*32+(lane&15) and +16;
  // VGPRs 0-7: K=16*hi+{0..15}
  const _Float16* Ap0 = A + (size_t)(mt*32 + (lane & 15))*lda + (hi << 4);
  const _Float16* Ap1 = Ap0 + (size_t)16*lda;
  // B fragments: lane holds col nt*64+j*16+(lane&15); VGPRs 0-7: K=16*hi+{0..15}
  int ncol = nt*64 + (lane & 15);
  const _Float16* Bp0 = Bt + (size_t)(ncol     )*K + (hi << 4);
  const _Float16* Bp1 = Bt + (size_t)(ncol + 16)*K + (hi << 4);
  const _Float16* Bp2 = Bt + (size_t)(ncol + 32)*K + (hi << 4);
  const _Float16* Bp3 = Bt + (size_t)(ncol + 48)*K + (hi << 4);
  v8f acc00 = {}, acc01 = {}, acc02 = {}, acc03 = {};
  v8f acc10 = {}, acc11 = {}, acc12 = {}, acc13 = {};
  for (int k0 = 0; k0 < K; k0 += 32) {
    __builtin_prefetch(Ap0 + k0 + 64, 0, 1);
    __builtin_prefetch(Bp0 + k0 + 64, 0, 1);
    v16h a0, a1;
    *(v8h*)&a0       = *(const v8h*)(Ap0 + k0);
    *((v8h*)&a0 + 1) = *(const v8h*)(Ap0 + k0 + 8);
    *(v8h*)&a1       = *(const v8h*)(Ap1 + k0);
    *((v8h*)&a1 + 1) = *(const v8h*)(Ap1 + k0 + 8);
    v16h b0 = *(const v16h*)(Bp0 + k0);
    v16h b1 = *(const v16h*)(Bp1 + k0);
    v16h b2 = *(const v16h*)(Bp2 + k0);
    v16h b3 = *(const v16h*)(Bp3 + k0);
    acc00 = __builtin_amdgcn_wmma_f32_16x16x32_f16(false, a0, false, b0, (short)0, acc00, false, false);
    acc10 = __builtin_amdgcn_wmma_f32_16x16x32_f16(false, a1, false, b0, (short)0, acc10, false, false);
    acc01 = __builtin_amdgcn_wmma_f32_16x16x32_f16(false, a0, false, b1, (short)0, acc01, false, false);
    acc11 = __builtin_amdgcn_wmma_f32_16x16x32_f16(false, a1, false, b1, (short)0, acc11, false, false);
    acc02 = __builtin_amdgcn_wmma_f32_16x16x32_f16(false, a0, false, b2, (short)0, acc02, false, false);
    acc12 = __builtin_amdgcn_wmma_f32_16x16x32_f16(false, a1, false, b2, (short)0, acc12, false, false);
    acc03 = __builtin_amdgcn_wmma_f32_16x16x32_f16(false, a0, false, b3, (short)0, acc03, false, false);
    acc13 = __builtin_amdgcn_wmma_f32_16x16x32_f16(false, a1, false, b3, (short)0, acc13, false, false);
  }
  // C/D layout: VGPR v -> M = tileRow + v + 8*hi ; N = strip + (lane&15)
  int nb = nt*64 + (lane & 15);
  v8f accs[2][4] = {{acc00, acc01, acc02, acc03}, {acc10, acc11, acc12, acc13}};
  #pragma unroll
  for (int mi = 0; mi < 2; ++mi) {
    int mbase = mt*32 + mi*16 + (hi << 3);
    #pragma unroll
    for (int j = 0; j < 4; ++j) {
      int n = nb + j*16;
      if (n >= Nout) continue;
      #pragma unroll
      for (int v = 0; v < 8; ++v) {
        size_t idx = (size_t)(mbase + v)*ldc + n;
        float val = accs[mi][j][v];
        if (EPI == 0)      C[idx] = val;
        else if (EPI == 1) C[idx] = C[idx] + val;
        else if (EPI == 2) C[idx] = val + aux[n];
        else { float g = aux[idx]; float s = g/(1.0f + __expf(-g)); C16[idx] = (_Float16)(s*val); }
      }
    }
  }
}

// ---------------------------------------------------------------- sliding-window causal attention, wave per (b,s,h)
// qkv fused buffer [MROWS, 1536]: q at col 0, k at 512, v at 1024
__global__ void k_attn(const float* __restrict__ qkv, _Float16* __restrict__ o) {
  int w = blockIdx.x*(blockDim.x >> 5) + (threadIdx.x >> 5);
  if (w >= MROWS*NHEAD) return;
  int m = w >> 3, hh = w & 7;
  int s = m % SLEN;
  int lane = threadIdx.x & 31;
  const float* qp = qkv + (size_t)m*LDQKV + hh*DHEAD;
  float q0 = qp[lane], q1 = qp[lane + 32];
  int jmax = (s + 1 < SWINDOW) ? (s + 1) : SWINDOW;
  float sc[SWINDOW];
  #pragma unroll
  for (int j = 0; j < SWINDOW; ++j) {
    float d = -1e30f;
    if (j < jmax) {                       // wave-uniform branch
      const float* kp = qkv + (size_t)(m - j)*LDQKV + D_MODEL + hh*DHEAD;
      d = q0*kp[lane] + q1*kp[lane + 32];
      #pragma unroll
      for (int off = 16; off > 0; off >>= 1) d += __shfl_xor(d, off, 32);
      d *= 0.125f;                        // 1/sqrt(64)
    }
    sc[j] = d;
  }
  float mx = -1e30f;
  #pragma unroll
  for (int j = 0; j < SWINDOW; ++j) mx = fmaxf(mx, sc[j]);
  float den = 0.0f, o0 = 0.0f, o1 = 0.0f;
  #pragma unroll
  for (int j = 0; j < SWINDOW; ++j) {
    if (j < jmax) {
      float p = __expf(sc[j] - mx);
      den += p;
      const float* vp = qkv + (size_t)(m - j)*LDQKV + 2*D_MODEL + hh*DHEAD;
      o0 += p*vp[lane]; o1 += p*vp[lane + 32];
    }
  }
  float inv = 1.0f/den;
  _Float16* op = o + (size_t)m*LDH + hh*DHEAD;
  op[lane]      = (_Float16)(o0*inv);
  op[lane + 32] = (_Float16)(o1*inv);
}

// ---------------------------------------------------------------- spectral features: y[.,1282] -> f16 [Re_0..640 | Im_0..640 | pad]
__global__ void k_feat(const float* __restrict__ y, _Float16* __restrict__ f) {
  int idx = blockIdx.x*blockDim.x + threadIdx.x;
  if (idx >= MROWS*KFFT) return;
  int m = idx / KFFT, c = idx - m*KFFT;
  float val = 0.0f;
  if (c < NBINS) {
    float la = y[(size_t)m*YC + c], an = y[(size_t)m*YC + NBINS + c];
    val = __expf(la)*cosf(an);
  } else if (c < YC) {
    int k = c - NBINS;
    float la = y[(size_t)m*YC + k], an = y[(size_t)m*YC + NBINS + k];
    val = __expf(la)*sinf(an);
  }
  f[(size_t)m*LDH + c] = (_Float16)val;
}

// ---------------------------------------------------------------- overlap-add + env normalization (gather, no atomics)
__global__ void k_ola(const float* __restrict__ frames, float* __restrict__ out) {
  int i = blockIdx.x*blockDim.x + threadIdx.x;
  if (i >= BATCH*OUTLEN) return;
  int b = i / OUTLEN, ii = i - b*OUTLEN;
  int p = ii + PADC;
  int smax = p / HOPC; if (smax > SLEN - 1) smax = SLEN - 1;
  int smin = (p - (NFFTC - 1) + (HOPC - 1)) / HOPC; if (smin < 0) smin = 0;
  float acc = 0.0f, env = 0.0f;
  const float w0 = 6.28318530717958647692f/(float)NFFTC;
  for (int s2 = smin; s2 <= smax; ++s2) {
    int n = p - s2*HOPC;
    acc += frames[((size_t)b*SLEN + s2)*NFFTC + n];
    float wv = 0.5f*(1.0f - cosf(w0*(float)n));
    env += wv*wv;
  }
  out[i] = acc/env;
}

// ================================================================ launch
extern "C" void kernel_launch(void* const* d_in, const int* in_sizes, int n_in,
                              void* d_out, int out_size, void* d_ws, size_t ws_size,
                              hipStream_t stream) {
  (void)in_sizes; (void)n_in; (void)out_size; (void)ws_size;
  const int*   codes = (const int*)  d_in[0];
  const float* emb   = (const float*)d_in[1];
  const float* ln1   = (const float*)d_in[2];
  const float* ln2   = (const float*)d_in[3];
  const float* fln   = (const float*)d_in[4];
  const float* wq    = (const float*)d_in[5];
  const float* wk    = (const float*)d_in[6];
  const float* wv    = (const float*)d_in[7];
  const float* wo    = (const float*)d_in[8];
  const float* wg    = (const float*)d_in[9];
  const float* wu    = (const float*)d_in[10];
  const float* wd    = (const float*)d_in[11];
  const float* linw  = (const float*)d_in[12];
  const float* linb  = (const float*)d_in[13];
  char* ws = (char*)d_ws;

  // workspace layout (bytes): needs ~111 MB
  const size_t LSH   = 4194304;                      // f16 weight halves per layer
  const size_t oW    = 0;
  const size_t oTbl  = 34930688;                     // (4*LSH + 1344*512)*2
  const size_t oH    = oTbl + (size_t)NFFTC*KFFT*2;  // f16 act buf A
  const size_t oH2   = oH  + (size_t)MROWS*LDH*2;    // f16 act buf B (MLP t)
  const size_t oX    = oH2 + (size_t)MROWS*LDH*2;    // f32 residual x
  const size_t oBig  = oX  + (size_t)MROWS*D_MODEL*4;// f32 scratch region

  _Float16* W16 = (_Float16*)(ws + oW);
  _Float16* tbl = (_Float16*)(ws + oTbl);
  _Float16* hb  = (_Float16*)(ws + oH);
  _Float16* hb2 = (_Float16*)(ws + oH2);
  float*    xb  = (float*)(ws + oX);
  float*    big = (float*)(ws + oBig);
  float* qkvb = big;                                 // [3072,1536] fused q|k|v
  float* gb   = big;                                 // reused after attention
  float* yb   = big;                                 // reused after last layer
  float* fr   = big + (size_t)MROWS*FFDIM;           // frames, after g dead
  _Float16* lin_t = W16 + 4*LSH;

  const int TPB = 256;
  auto cvt = [&](const float* s, _Float16* d, int K, int N, int Np) {
    int tot = Np*K;
    k_convert_t<<<(tot + TPB - 1)/TPB, TPB, 0, stream>>>(s, d, K, N, Np);
  };
  for (int l = 0; l < NLAYER; ++l) {
    _Float16* base = W16 + (size_t)l*LSH;
    cvt(wq + (size_t)l*512*512,  base + 0,       512, 512, 512);   // q|k|v rows
    cvt(wk + (size_t)l*512*512,  base + 262144,  512, 512, 512);   //  contiguous ->
    cvt(wv + (size_t)l*512*512,  base + 524288,  512, 512, 512);   //  fused [1536,512]
    cvt(wo + (size_t)l*512*512,  base + 786432,  512, 512, 512);
    cvt(wg + (size_t)l*512*2048, base + 1048576, 512, 2048, 2048);
    cvt(wu + (size_t)l*512*2048, base + 2097152, 512, 2048, 2048);
    cvt(wd + (size_t)l*2048*512, base + 3145728, 2048, 512, 512);
  }
  cvt(linw, lin_t, 512, YC, NLIN);
  k_build_tbl<<<(NFFTC*KFFT + TPB - 1)/TPB, TPB, 0, stream>>>(tbl);

  k_embed<<<BATCH*TLEN, 256, 0, stream>>>(codes, emb, xb);

  const int MT = MROWS/32;                           // 96 M tiles of 32 rows
  auto blks = [&](int nTiles) { return (MT*nTiles + 3)/4; };

  for (int l = 0; l < NLAYER; ++l) {
    _Float16* base = W16 + (size_t)l*LSH;
    k_rmsnorm<<<MROWS/8, 256, 0, stream>>>(xb, ln1 + (size_t)l*512, hb);
    // fused QKV: B = [1536,512] (wq_t|wk_t|wv_t), C = [3072,1536]
    k_gemm<0><<<blks(24), 128, 0, stream>>>(hb, base + 0, qkvb, nullptr, nullptr,
                                            512, LDH, 24, LDQKV, LDQKV);
    k_attn<<<(MROWS*NHEAD + 3)/4, 128, 0, stream>>>(qkvb, hb);     // o -> hb (h dead)
    k_gemm<1><<<blks(8), 128, 0, stream>>>(hb, base + 786432, xb, nullptr, nullptr,
                                           512, LDH, 8, 512, 512);
    k_rmsnorm<<<MROWS/8, 256, 0, stream>>>(xb, ln2 + (size_t)l*512, hb);
    k_gemm<0><<<blks(32), 128, 0, stream>>>(hb, base + 1048576, gb, nullptr, nullptr,
                                            512, LDH, 32, 2048, 2048);
    k_gemm<3><<<blks(32), 128, 0, stream>>>(hb, base + 2097152, nullptr, hb2, gb,
                                            512, LDH, 32, 2048, 2048);
    k_gemm<1><<<blks(8), 128, 0, stream>>>(hb2, base + 3145728, xb, nullptr, nullptr,
                                           2048, LDH, 8, 512, 512);
  }

  k_rmsnorm<<<MROWS/8, 256, 0, stream>>>(xb, fln, hb);
  k_gemm<2><<<blks(NLIN/64), 128, 0, stream>>>(hb, lin_t, yb, nullptr, linb,
                                               512, LDH, NLIN/64, YC, YC);
  k_feat<<<(MROWS*KFFT + TPB - 1)/TPB, TPB, 0, stream>>>(yb, hb);
  k_gemm<0><<<blks(NFFTC/64), 128, 0, stream>>>(hb, tbl, fr, nullptr, nullptr,
                                                KFFT, LDH, NFFTC/64, NFFTC, NFFTC);
  k_ola<<<(BATCH*OUTLEN + TPB - 1)/TPB, TPB, 0, stream>>>(fr, (float*)d_out);
}
